// CovenantAttention_17635135717476
// MI455X (gfx1250) — matrix operand order repaired and checked by reading, not compiled, not run on hardware
//
#include <hip/hip_runtime.h>
#include <hip/hip_bf16.h>
#include <math.h>

// ---------------------------------------------------------------------------
// CovenantAttention for MI455X (gfx1250, wave32, WMMA).
// - All GEMM stages: v_wmma_f32_16x16x32_bf16 (K=32/instr, f32 accum)
// - Complex mults use a pre-negated imag-weight copy so each complex output
//   needs only 2 accumulators (no B-negate modifier on bf16 WMMA).
// - Attention: block-shared K/V staged via global_load_async_to_lds_b128,
//   double-buffered with s_wait_asynccnt.
// ---------------------------------------------------------------------------

typedef __bf16 v16bf __attribute__((ext_vector_type(16)));
typedef __bf16 v8bf  __attribute__((ext_vector_type(8)));
typedef float  v8f   __attribute__((ext_vector_type(8)));

#define D_MODEL 1024
#define SEQ     2048
#define BATCH   2
#define NROWS   (BATCH * SEQ)   // 4096
#define HEADS   8
#define HD      128

__device__ __forceinline__ v8f zero8() {
    v8f r;
#pragma unroll
    for (int j = 0; j < 8; ++j) r[j] = 0.0f;
    return r;
}

// A-fragment (16x32 bf16, MxK). ISA 7.12.2: lane m (0..15) rows; lane half
// selects K-octet. Per lane: K = half*8+{0..7} and 16+half*8+{0..7}.
__device__ __forceinline__ v16bf load_fragA(const __bf16* base, int ld,
                                            int row0, int k0, int lane) {
    int m = lane & 15, half = lane >> 4;
    const __bf16* p = base + (size_t)(row0 + m) * ld + k0 + half * 8;
    v8bf lo = *(const v8bf*)p;          // K = half*8 + 0..7
    v8bf hi = *(const v8bf*)(p + 16);   // K = 16 + half*8 + 0..7
    v16bf r;
#pragma unroll
    for (int j = 0; j < 8; ++j) { r[j] = lo[j]; r[j + 8] = hi[j]; }
    return r;
}

// B-fragment (32x16 bf16, KxN). Lane n = lane&15 owns column n; per lane:
// K = half*8+{0..7} and 16+half*8+{0..7}. `base` indexed [col][k].
__device__ __forceinline__ v16bf load_fragB(const __bf16* base, int ld,
                                            int col0, int k0, int lane) {
    int n = lane & 15, half = lane >> 4;
    const __bf16* p = base + (size_t)(col0 + n) * ld + k0 + half * 8;
    v8bf a = *(const v8bf*)p;
    v8bf b = *(const v8bf*)(p + 16);
    v16bf r;
#pragma unroll
    for (int j = 0; j < 8; ++j) { r[j] = a[j]; r[j + 8] = b[j]; }
    return r;
}

__device__ __forceinline__ v8f wmma_bf16(v16bf a, v16bf b, v8f c) {
    return __builtin_amdgcn_wmma_f32_16x16x32_bf16(
        false, a, false, b, (short)0, c, false, false);
}

// gfx1250 async copy: per-lane 16B global -> LDS, tracked by ASYNCcnt.
// LDS byte address = low 32 bits of the generic pointer (flat-LDS aperture).
__device__ __forceinline__ void async_copy16(const __bf16* gsrc, const __bf16* lds) {
    unsigned lofs = (unsigned)(size_t)lds;
    unsigned long long gaddr = (unsigned long long)(size_t)gsrc;
    asm volatile("global_load_async_to_lds_b128 %0, %1, off"
                 :: "v"(lofs), "v"(gaddr) : "memory");
}

// ---------------------------------------------------------------------------
// fp32 -> bf16 streaming converts (HBM-bound, one pass)
// ---------------------------------------------------------------------------
__global__ void k_f32_to_bf16(const float* __restrict__ src,
                              __bf16* __restrict__ dst, int n) {
    int i = blockIdx.x * blockDim.x + threadIdx.x;
    if (i < n) dst[i] = (__bf16)src[i];
}

// writes value and its negation (for the 2-accumulator complex GEMM trick)
__global__ void k_f32_to_bf16_pm(const float* __restrict__ src,
                                 __bf16* __restrict__ dst,
                                 __bf16* __restrict__ nds, int n) {
    int i = blockIdx.x * blockDim.x + threadIdx.x;
    if (i < n) { float v = src[i]; dst[i] = (__bf16)v; nds[i] = (__bf16)(-v); }
}

// ---------------------------------------------------------------------------
// Zeta basis, generated transposed: Zt[j][d] = Z[d][j]; also negated imag.
// ---------------------------------------------------------------------------
__global__ void k_zeta(__bf16* __restrict__ ztr, __bf16* __restrict__ zti,
                       __bf16* __restrict__ ztn) {
    int i = blockIdx.x * blockDim.x + threadIdx.x;
    if (i >= HD * HD) return;
    int j = i >> 7;
    int d = i & 127;
    float t   = 14.134725f + 9.0f * (float)j;
    float ln  = logf((float)(d + 1));
    float amp = rsqrtf((float)(d + 1)) * 0.08838834764831845f; // 1/sqrt(128)
    float ph  = t * ln;
    float sv  = amp * sinf(ph);
    ztr[i] = (__bf16)(amp * cosf(ph));
    zti[i] = (__bf16)(-sv);
    ztn[i] = (__bf16)( sv);
}

// ---------------------------------------------------------------------------
// Complex GEMM: C = A(complex [4096 x 1024] bf16) @ W^T(complex) + bias.
// 32x32 complex tile per wave, 2 accumulators/tile via pre-negated Wn = -Wi:
//   Cre += ar*Wr + ai*Wn ;  Cim += ar*Wi + ai*Wr
// mode 0: bf16 [row][col];  mode 1: bf16 V^T [b][dim][s];  mode 2: complex64.
// ---------------------------------------------------------------------------
__global__ void __launch_bounds__(256, 1)
k_cgemm(const __bf16* __restrict__ Ar, const __bf16* __restrict__ Ai,
        const __bf16* __restrict__ Wr, const __bf16* __restrict__ Wi,
        const __bf16* __restrict__ Wn,
        const float* __restrict__ br, const float* __restrict__ bi,
        __bf16* __restrict__ Cr, __bf16* __restrict__ Ci,
        float2* __restrict__ Cout, int mode) {
    int lane = threadIdx.x & 31;
    int w = blockIdx.x * (blockDim.x >> 5) + (threadIdx.x >> 5);
    const int tilesN = D_MODEL / 32;           // 32
    int mt = (w / tilesN) * 32;
    int nt = (w % tilesN) * 32;

    v8f acc_re[2][2], acc_im[2][2];
#pragma unroll
    for (int mi = 0; mi < 2; ++mi)
#pragma unroll
        for (int ni = 0; ni < 2; ++ni) { acc_re[mi][ni] = zero8(); acc_im[mi][ni] = zero8(); }

    for (int k0 = 0; k0 < D_MODEL; k0 += 32) {
        v16bf ar[2], ai[2], fr[2], fi[2], fn[2];
#pragma unroll
        for (int mi = 0; mi < 2; ++mi) {
            ar[mi] = load_fragA(Ar, D_MODEL, mt + mi * 16, k0, lane);
            ai[mi] = load_fragA(Ai, D_MODEL, mt + mi * 16, k0, lane);
        }
#pragma unroll
        for (int ni = 0; ni < 2; ++ni) {
            fr[ni] = load_fragB(Wr, D_MODEL, nt + ni * 16, k0, lane);
            fi[ni] = load_fragB(Wi, D_MODEL, nt + ni * 16, k0, lane);
            fn[ni] = load_fragB(Wn, D_MODEL, nt + ni * 16, k0, lane);
        }
#pragma unroll
        for (int mi = 0; mi < 2; ++mi)
#pragma unroll
            for (int ni = 0; ni < 2; ++ni) {
                acc_re[mi][ni] = wmma_bf16(ar[mi], fr[ni], acc_re[mi][ni]);
                acc_re[mi][ni] = wmma_bf16(ai[mi], fn[ni], acc_re[mi][ni]);
                acc_im[mi][ni] = wmma_bf16(ar[mi], fi[ni], acc_im[mi][ni]);
                acc_im[mi][ni] = wmma_bf16(ai[mi], fr[ni], acc_im[mi][ni]);
            }
    }

    int nlane = lane & 15, half = lane >> 4;
#pragma unroll
    for (int ni = 0; ni < 2; ++ni) {
        int col = nt + ni * 16 + nlane;
        float bre = br[col], bim = bi[col];
#pragma unroll
        for (int mi = 0; mi < 2; ++mi)
#pragma unroll
            for (int r = 0; r < 8; ++r) {
                int row = mt + mi * 16 + r + half * 8;
                float cre = acc_re[mi][ni][r] + bre;
                float cim = acc_im[mi][ni][r] + bim;
                if (mode == 0) {
                    Cr[(size_t)row * D_MODEL + col] = (__bf16)cre;
                    Ci[(size_t)row * D_MODEL + col] = (__bf16)cim;
                } else if (mode == 1) {            // V^T : [b][dim][s]
                    int b = row >> 11, s = row & (SEQ - 1);
                    size_t idx = ((size_t)b * D_MODEL + col) * SEQ + s;
                    Cr[idx] = (__bf16)cre;
                    Ci[idx] = (__bf16)cim;
                } else {
                    Cout[(size_t)row * D_MODEL + col] = make_float2(cre, cim);
                }
            }
    }
}

// ---------------------------------------------------------------------------
// Per-head zeta projection (block-diagonal complex GEMM, K=128).
// Optional negated-imag output (for K, consumed by attention scores).
// ---------------------------------------------------------------------------
__global__ void __launch_bounds__(256, 1)
k_zgemm(const __bf16* __restrict__ Pr, const __bf16* __restrict__ Pi,
        const __bf16* __restrict__ Ztr, const __bf16* __restrict__ Zti,
        const __bf16* __restrict__ Ztn,
        __bf16* __restrict__ Or, __bf16* __restrict__ Oi,
        __bf16* __restrict__ Oin, float scale) {
    int lane = threadIdx.x & 31;
    int w = blockIdx.x * (blockDim.x >> 5) + (threadIdx.x >> 5);
    const int tilesN = D_MODEL / 16;
    int mt = (w / tilesN) * 16;
    int nt = (w % tilesN) * 16;
    int h  = nt >> 7;
    int j0 = nt & (HD - 1);

    v8f acc_re = zero8(), acc_im = zero8();
#pragma unroll
    for (int d0 = 0; d0 < HD; d0 += 32) {
        v16bf ar = load_fragA(Pr, D_MODEL, mt, h * HD + d0, lane);
        v16bf ai = load_fragA(Pi, D_MODEL, mt, h * HD + d0, lane);
        v16bf zr = load_fragB(Ztr, HD, j0, d0, lane);
        v16bf zi = load_fragB(Zti, HD, j0, d0, lane);
        v16bf zn = load_fragB(Ztn, HD, j0, d0, lane);
        acc_re = wmma_bf16(ar, zr, acc_re);
        acc_re = wmma_bf16(ai, zn, acc_re);
        acc_im = wmma_bf16(ar, zi, acc_im);
        acc_im = wmma_bf16(ai, zr, acc_im);
    }
    int col = nt + (lane & 15);
    int half = lane >> 4;
#pragma unroll
    for (int r = 0; r < 8; ++r) {
        int row = mt + r + half * 8;
        float vre = acc_re[r] * scale, vim = acc_im[r] * scale;
        Or[(size_t)row * D_MODEL + col] = (__bf16)vre;
        Oi[(size_t)row * D_MODEL + col] = (__bf16)vim;
        if (Oin) Oin[(size_t)row * D_MODEL + col] = (__bf16)(-vim);
    }
}

// ---------------------------------------------------------------------------
// Flash attention over |complex scores|, complex V.
// Block = 8 waves on the SAME (b,h); K/V staged in LDS by
// global_load_async_to_lds_b128, double-buffered (ASYNCcnt + barriers).
// LDS per buffer (40KB): Kr[32k][128d] | Ki | Kn(= -Ki) | Vr[128d][32k] | Vi
// Scores use 2 accumulators: Sr = Qr*Kr + Qi*Kn ; Si = Qr*Ki + Qi*Kr.
// ---------------------------------------------------------------------------
__global__ void __launch_bounds__(256, 1)
k_attn(const __bf16* __restrict__ Qr, const __bf16* __restrict__ Qi,
       const __bf16* __restrict__ Kr, const __bf16* __restrict__ Ki,
       const __bf16* __restrict__ Kn,
       const __bf16* __restrict__ Vtr, const __bf16* __restrict__ Vti,
       __bf16* __restrict__ Or, __bf16* __restrict__ Oi) {
    __shared__ __bf16 ldsKV[2 * 5 * 4096];   // 2 buffers * 5 tiles * 8KB = 80KB
    __shared__ __bf16 ldsP[8 * 512];         // per-wave 16x32 P transpose patch

    int lane = threadIdx.x & 31;
    int wid  = threadIdx.x >> 5;
    int bh = blockIdx.x >> 4;                // 16 blocks per (b,h)
    int qt = (blockIdx.x & 15) * 8 + wid;    // this wave's query tile
    int b = bh >> 3, h = bh & 7;
    int qrow = b * SEQ + qt * 16;
    __bf16* myP = ldsP + wid * 512;

    int half = lane >> 4, nlane = lane & 15;

    const __bf16* Vtb_r = Vtr + (size_t)b * D_MODEL * SEQ;
    const __bf16* Vtb_i = Vti + (size_t)b * D_MODEL * SEQ;

    // Issue this wave's 1/8 share of the 5-tile batch (10 async b128 ops).
    // 2560 chunks of 16B: t = tile (Kr,Ki,Kn,Vr,Vi), c = chunk within tile.
    auto issue_tiles = [&](int kt, int buf) {
#pragma unroll
        for (int i = 0; i < 10; ++i) {
            int L = wid * 320 + i * 32 + lane;
            int t = L >> 9, c = L & 511;
            const __bf16* g;
            if (t < 3) {                       // K tiles: [key][128 dims]
                const __bf16* Kb = (t == 0) ? Kr : ((t == 1) ? Ki : Kn);
                int key = c >> 4, off = c & 15;
                g = Kb + ((size_t)(b * SEQ + kt + key) * D_MODEL + h * HD) + off * 8;
            } else {                           // V tiles: [128 dims][32 keys]
                const __bf16* Vb = (t == 3) ? Vtb_r : Vtb_i;
                int dim = c >> 2, off = c & 3;
                g = Vb + ((size_t)(h * HD + dim) * SEQ + kt) + off * 8;
            }
            const __bf16* l = ldsKV + buf * 20480 + t * 4096 + c * 8;
            async_copy16(g, l);
        }
    };

    v8f o_r[8], o_i[8];
#pragma unroll
    for (int t = 0; t < 8; ++t) { o_r[t] = zero8(); o_i[t] = zero8(); }
    float mrun[8], lrun[8];
#pragma unroll
    for (int r = 0; r < 8; ++r) { mrun[r] = -3.0e38f; lrun[r] = 0.0f; }

    issue_tiles(0, 0);                         // prologue: fill buffer 0

    for (int kt = 0; kt < SEQ; kt += 32) {
        int buf = (kt >> 5) & 1;
        // Prefetch next tile batch, then wait for current batch to land.
        if (kt + 32 < SEQ) {
            issue_tiles(kt + 32, buf ^ 1);
            asm volatile("s_wait_asynccnt 10" ::: "memory");
        } else {
            asm volatile("s_wait_asynccnt 0" ::: "memory");
        }
        __syncthreads();                       // publish LDS to all 8 waves

        const __bf16* bK_r = ldsKV + buf * 20480;
        const __bf16* bK_i = bK_r + 4096;
        const __bf16* bK_n = bK_r + 8192;
        const __bf16* bV_r = bK_r + 12288;
        const __bf16* bV_i = bK_r + 16384;

        float mag[2][8];
#pragma unroll
        for (int c = 0; c < 2; ++c) {
            v8f sr = zero8(), si = zero8();
#pragma unroll
            for (int ks = 0; ks < 4; ++ks) {
                // Q fragments come from L0/L2 (same addresses every iter).
                v16bf qar = load_fragA(Qr, D_MODEL, qrow, h * HD + ks * 32, lane);
                v16bf qai = load_fragA(Qi, D_MODEL, qrow, h * HD + ks * 32, lane);
                v16bf kbr = load_fragB(bK_r, HD, c * 16, ks * 32, lane);
                v16bf kbi = load_fragB(bK_i, HD, c * 16, ks * 32, lane);
                v16bf kbn = load_fragB(bK_n, HD, c * 16, ks * 32, lane);
                sr = wmma_bf16(qar, kbr, sr);
                sr = wmma_bf16(qai, kbn, sr);
                si = wmma_bf16(qar, kbi, si);
                si = wmma_bf16(qai, kbr, si);
            }
#pragma unroll
            for (int r = 0; r < 8; ++r)
                mag[c][r] = sqrtf(sr[r] * sr[r] + si[r] * si[r]);
        }
        // Online softmax of |scores| (rows live across 16-lane halves).
        float alpha[8], pv[2][8];
#pragma unroll
        for (int r = 0; r < 8; ++r) {
            float mx = fmaxf(mag[0][r], mag[1][r]);
            for (int s = 1; s < 16; s <<= 1) mx = fmaxf(mx, __shfl_xor(mx, s, 32));
            float mnew = fmaxf(mrun[r], mx);
            alpha[r] = __expf(mrun[r] - mnew);
            float p0 = __expf(mag[0][r] - mnew);
            float p1 = __expf(mag[1][r] - mnew);
            pv[0][r] = p0; pv[1][r] = p1;
            float rs = p0 + p1;
            for (int s = 1; s < 16; s <<= 1) rs += __shfl_xor(rs, s, 32);
            lrun[r] = lrun[r] * alpha[r] + rs;
            mrun[r] = mnew;
        }
#pragma unroll
        for (int t = 0; t < 8; ++t)
#pragma unroll
            for (int r = 0; r < 8; ++r) { o_r[t][r] *= alpha[r]; o_i[t][r] *= alpha[r]; }

        // P (C-layout) -> row-major 16x32 bf16 in this wave's LDS patch.
#pragma unroll
        for (int c = 0; c < 2; ++c)
#pragma unroll
            for (int r = 0; r < 8; ++r)
                myP[(r + half * 8) * 32 + c * 16 + nlane] = (__bf16)pv[c][r];
        asm volatile("s_wait_dscnt 0" ::: "memory");
        v16bf pa = load_fragA(myP, 32, 0, 0, lane);

        // out += P @ V (complex V => two real WMMAs per 16-dim tile)
#pragma unroll
        for (int t = 0; t < 8; ++t) {
            v16bf vbr = load_fragB(bV_r, 32, t * 16, 0, lane);
            v16bf vbi = load_fragB(bV_i, 32, t * 16, 0, lane);
            o_r[t] = wmma_bf16(pa, vbr, o_r[t]);
            o_i[t] = wmma_bf16(pa, vbi, o_i[t]);
        }
        __syncthreads();          // all waves done reading before buf refill
    }

#pragma unroll
    for (int r = 0; r < 8; ++r) {
        float inv = 1.0f / lrun[r];
        int row = qrow + r + half * 8;
#pragma unroll
        for (int t = 0; t < 8; ++t) {
            int col = h * HD + t * 16 + nlane;
            Or[(size_t)row * D_MODEL + col] = (__bf16)(o_r[t][r] * inv);
            Oi[(size_t)row * D_MODEL + col] = (__bf16)(o_i[t][r] * inv);
        }
    }
}

// ---------------------------------------------------------------------------
extern "C" void kernel_launch(void* const* d_in, const int* in_sizes, int n_in,
                              void* d_out, int out_size, void* d_ws, size_t ws_size,
                              hipStream_t stream) {
    const float* x_real = (const float*)d_in[0];
    const float* x_imag = (const float*)d_in[1];
    const float* W_r    = (const float*)d_in[2];
    const float* W_i    = (const float*)d_in[3];
    const float* b_r    = (const float*)d_in[4];
    const float* b_i    = (const float*)d_in[5];

    const size_t XE = (size_t)NROWS * D_MODEL;        // 4 Mi elems
    const size_t WE = (size_t)4 * D_MODEL * D_MODEL;  // 4 Mi elems
    const size_t ZE = (size_t)HD * HD;

    char* base = (char*)d_ws;
    size_t off = 0;
    auto alloc = [&](size_t elems) -> __bf16* {
        __bf16* p = (__bf16*)(base + off);
        off += ((elems * sizeof(__bf16)) + 255) & ~(size_t)255;
        return p;
    };
    __bf16* Xr   = alloc(XE);  __bf16* Xi   = alloc(XE);
    __bf16* Wbr  = alloc(WE);  __bf16* Wbi  = alloc(WE);  __bf16* Wbn = alloc(WE);
    __bf16* Ztr  = alloc(ZE);  __bf16* Zti  = alloc(ZE);  __bf16* Ztn = alloc(ZE);
    __bf16* P0r  = alloc(XE);  __bf16* P0i  = alloc(XE);
    __bf16* P1r  = alloc(XE);  __bf16* P1i  = alloc(XE);
    __bf16* Qzr  = alloc(XE);  __bf16* Qzi  = alloc(XE);
    __bf16* Kzr  = alloc(XE);  __bf16* Kzi  = alloc(XE);  __bf16* Kzn = alloc(XE);
    __bf16* Vtr  = alloc(XE);  __bf16* Vti  = alloc(XE);
    __bf16* O1r  = P0r;        __bf16* O1i  = P0i;    // reuse: P0 dead after zeta

    dim3 blk(256);
    const size_t WSTRIDE = (size_t)D_MODEL * D_MODEL;

    k_f32_to_bf16<<<dim3((unsigned)((XE + 255) / 256)), blk, 0, stream>>>(x_real, Xr, (int)XE);
    k_f32_to_bf16<<<dim3((unsigned)((XE + 255) / 256)), blk, 0, stream>>>(x_imag, Xi, (int)XE);
    k_f32_to_bf16<<<dim3((unsigned)((WE + 255) / 256)), blk, 0, stream>>>(W_r, Wbr, (int)WE);
    k_f32_to_bf16_pm<<<dim3((unsigned)((WE + 255) / 256)), blk, 0, stream>>>(W_i, Wbi, Wbn, (int)WE);
    k_zeta<<<dim3((unsigned)((ZE + 255) / 256)), blk, 0, stream>>>(Ztr, Zti, Ztn);

    const int CG_BLOCKS = (NROWS / 32) * (D_MODEL / 32) / 8;     // 512
    const int ZG_BLOCKS = (NROWS / 16) * (D_MODEL / 16) / 8;     // 2048
    // Q,K pre-zeta projections
    k_cgemm<<<CG_BLOCKS, blk, 0, stream>>>(Xr, Xi, Wbr + 0 * WSTRIDE, Wbi + 0 * WSTRIDE,
                                           Wbn + 0 * WSTRIDE,
                                           b_r + 0 * D_MODEL, b_i + 0 * D_MODEL,
                                           P0r, P0i, nullptr, 0);
    k_cgemm<<<CG_BLOCKS, blk, 0, stream>>>(Xr, Xi, Wbr + 1 * WSTRIDE, Wbi + 1 * WSTRIDE,
                                           Wbn + 1 * WSTRIDE,
                                           b_r + 1 * D_MODEL, b_i + 1 * D_MODEL,
                                           P1r, P1i, nullptr, 0);
    // V projection, stored transposed [b][dim][s]
    k_cgemm<<<CG_BLOCKS, blk, 0, stream>>>(Xr, Xi, Wbr + 2 * WSTRIDE, Wbi + 2 * WSTRIDE,
                                           Wbn + 2 * WSTRIDE,
                                           b_r + 2 * D_MODEL, b_i + 2 * D_MODEL,
                                           Vtr, Vti, nullptr, 1);
    // Zeta projections (scale 1/sqrt(HD) folded into Q; K also writes -imag)
    k_zgemm<<<ZG_BLOCKS, blk, 0, stream>>>(P0r, P0i, Ztr, Zti, Ztn,
                                           Qzr, Qzi, nullptr, 0.08838834764831845f);
    k_zgemm<<<ZG_BLOCKS, blk, 0, stream>>>(P1r, P1i, Ztr, Zti, Ztn,
                                           Kzr, Kzi, Kzn, 1.0f);
    // Flash attention: 256 blocks, 8 waves each, block-shared K/V via async LDS
    const int ATT_BLOCKS = (BATCH * HEADS * (SEQ / 16)) / 8;     // 256
    k_attn<<<ATT_BLOCKS, blk, 0, stream>>>(Qzr, Qzi, Kzr, Kzi, Kzn, Vtr, Vti, O1r, O1i);
    // Output projection -> interleaved complex64
    k_cgemm<<<CG_BLOCKS, blk, 0, stream>>>(O1r, O1i, Wbr + 3 * WSTRIDE, Wbi + 3 * WSTRIDE,
                                           Wbn + 3 * WSTRIDE,
                                           b_r + 3 * D_MODEL, b_i + 3 * D_MODEL,
                                           nullptr, nullptr, (float2*)d_out, 2);
}
